// Decoder_61254823576144
// MI455X (gfx1250) — matrix operand order repaired and checked by reading, not compiled, not run on hardware
//
#include <hip/hip_runtime.h>

// ---------------------------------------------------------------------------
// Transformer decoder block for MI455X (gfx1250, wave32, WMMA bf16 16x16x32,
// TDM tensor_load_to_lds staging for the FFN GEMMs)
// ---------------------------------------------------------------------------

#define DEVI __device__ __forceinline__

typedef __attribute__((ext_vector_type(16))) __bf16 v16bf;
typedef __attribute__((ext_vector_type(8)))  __bf16 v8bf;
typedef __attribute__((ext_vector_type(4)))  __bf16 v4bf;
typedef __attribute__((ext_vector_type(8)))  float  v8f;
typedef __attribute__((ext_vector_type(4)))  unsigned int v4u;
typedef __attribute__((ext_vector_type(8)))  int v8i;
typedef __attribute__((ext_vector_type(4)))  int v4i;

constexpr int Bd  = 4;
constexpr int Td  = 1024;
constexpr int Cd  = 1024;
constexpr int Hh  = 16;
constexpr int HDd = 64;
constexpr int Fd  = 4096;          // 4*C
constexpr int Rows = Bd * Td;      // 4096

// ---------------- WMMA helpers ----------------

DEVI v8f wmma_bf16(v16bf a, v16bf b, v8f c) {
    // D = A(16x32 bf16) * B(32x16 bf16) + C(16x16 f32)
    return __builtin_amdgcn_wmma_f32_16x16x32_bf16(false, a, false, b, (short)0, c,
                                                   false, false);
}

// A operand (16x32, row-major source, stride lda elements):
// lane l<16: row=row0+l, K in {k0..k0+7, k0+16..k0+23}
// lane l>=16: row=row0+(l-16), K in {k0+8..k0+15, k0+24..k0+31}
template <typename P>
DEVI v16bf load_a_tile(P A, int lda, int row0, int k0, int lane) {
    int row  = row0 + (lane & 15);
    int koff = k0 + ((lane >> 4) << 3);
    auto* p = A + (size_t)row * lda + koff;
    union { v16bf v; v8bf h[2]; } u;
    u.h[0] = *(const v8bf*)(p);
    u.h[1] = *(const v8bf*)(p + 16);
    return u.v;
}

// B operand (32x16, row-major source [K,N], stride ldb elements):
// lane l holds column n0+(l&15), K in {k0+8h..+7, k0+16+8h..+7}, h=l>>4.
template <typename P>
DEVI v16bf load_b_tile(P Bm, int ldb, int k0, int n0, int lane) {
    auto* p = Bm + (size_t)(k0 + ((lane >> 4) << 3)) * ldb + n0 + (lane & 15);
    v16bf v;
#pragma unroll
    for (int s = 0; s < 8; s++) {
        v[s]     = p[(size_t)s * ldb];
        v[s + 8] = p[(size_t)(s + 16) * ldb];
    }
    return v;
}

// ---------------- TDM: async 2D tile DMA, global -> LDS ----------------
// Builds the CDNA5 Tensor DMA Descriptor (ISA ch.8) for a 2D bf16 tile whose
// tensor dims equal the tile dims (no OOB), dim0 stride = matrix row stride.
DEVI void tdm_load_2d(unsigned lds_off, const __bf16* gptr,
                      unsigned tile_w, unsigned tile_h, unsigned stride_elems) {
    unsigned lo = (unsigned)(size_t)gptr;
    unsigned hi = (unsigned)(((size_t)gptr) >> 32);
    v4u g0;
    g0[0] = 1u;                                   // count=1, user mode
    g0[1] = lds_off;                              // lds_addr (bytes)
    g0[2] = lo;                                   // global_addr[31:0]
    g0[3] = (hi & 0x1FFFFFFu) | (2u << 30);       // global_addr[56:32] | type=2
    v8i g1;
    g1[0] = (int)(1u << 16);                      // data_size=1 (2 bytes/elem)
    g1[1] = (int)((tile_w & 0xFFFFu) << 16);      // tensor_dim0[15:0]
    g1[2] = (int)((tile_w >> 16) | ((tile_h & 0xFFFFu) << 16)); // d0 hi | d1 lo
    g1[3] = (int)((tile_h >> 16) | (tile_w << 16));             // d1 hi | tile_dim0
    g1[4] = (int)tile_h;                          // tile_dim1 (tile_dim2=0)
    g1[5] = (int)stride_elems;                    // tensor_dim0_stride[31:0]
    g1[6] = 0;
    g1[7] = 0;
    v4i z4 = {0, 0, 0, 0};
    v8i z8 = {0, 0, 0, 0, 0, 0, 0, 0};
    // 6-arg form (clang-23 / therock-10.0 headers): groups 2,3 + aux group, cpol
    __builtin_amdgcn_tensor_load_to_lds(g0, g1, z4, z4, z8, 0);
}

// ---------------- fp32 -> bf16 conversion ----------------

__global__ __launch_bounds__(256) void cvt_bf16_kernel(const float* __restrict__ s,
                                                       __bf16* __restrict__ d) {
    size_t i = ((size_t)blockIdx.x * 256 + threadIdx.x) * 4;
    float4 v = *(const float4*)(s + i);
    v4bf o;
    o[0] = (__bf16)v.x; o[1] = (__bf16)v.y; o[2] = (__bf16)v.z; o[3] = (__bf16)v.w;
    *(v4bf*)(d + i) = o;
}

// ---------------- LayerNorm ----------------

DEVI void block_reduce2(float& a, float& b, float* sa, float* sb) {
    __syncthreads();
    int lane = threadIdx.x & 31, wv = threadIdx.x >> 5;
#pragma unroll
    for (int off = 16; off; off >>= 1) {
        a += __shfl_xor(a, off);
        b += __shfl_xor(b, off);
    }
    if (lane == 0) { sa[wv] = a; sb[wv] = b; }
    __syncthreads();
    if (wv == 0) {
        float ta = (lane < 8) ? sa[lane] : 0.f;
        float tb = (lane < 8) ? sb[lane] : 0.f;
#pragma unroll
        for (int off = 4; off; off >>= 1) {
            ta += __shfl_xor(ta, off);
            tb += __shfl_xor(tb, off);
        }
        if (lane == 0) { sa[0] = ta; sb[0] = tb; }
    }
    __syncthreads();
    a = sa[0]; b = sb[0];
}

__global__ __launch_bounds__(256) void ln_kernel(const float* __restrict__ x,
                                                 const float* __restrict__ g,
                                                 const float* __restrict__ bb,
                                                 float* __restrict__ of32,
                                                 __bf16* __restrict__ obf) {
    __shared__ float sa[8], sb[8];
    size_t row = blockIdx.x;
    int tid = threadIdx.x;
    float4 v = *(const float4*)(x + row * Cd + tid * 4);
    float s = v.x + v.y + v.z + v.w;
    float q = v.x * v.x + v.y * v.y + v.z * v.z + v.w * v.w;
    block_reduce2(s, q, sa, sb);
    float mu = s * (1.f / Cd);
    float rs = rsqrtf(q * (1.f / Cd) - mu * mu + 1e-5f);
    float4 gv = *(const float4*)(g + tid * 4);
    float4 bv = *(const float4*)(bb + tid * 4);
    float4 o;
    o.x = (v.x - mu) * rs * gv.x + bv.x;
    o.y = (v.y - mu) * rs * gv.y + bv.y;
    o.z = (v.z - mu) * rs * gv.z + bv.z;
    o.w = (v.w - mu) * rs * gv.w + bv.w;
    *(float4*)(of32 + row * Cd + tid * 4) = o;
    v4bf ob;
    ob[0] = (__bf16)o.x; ob[1] = (__bf16)o.y; ob[2] = (__bf16)o.z; ob[3] = (__bf16)o.w;
    *(v4bf*)(obf + row * Cd + tid * 4) = ob;
}

__global__ __launch_bounds__(256) void ln2f_kernel(const float* __restrict__ x2,
                                                   const float* __restrict__ g2,
                                                   const float* __restrict__ b2,
                                                   const float* __restrict__ gf,
                                                   const float* __restrict__ bfp,
                                                   float* __restrict__ x3,
                                                   float* __restrict__ hf32,
                                                   __bf16* __restrict__ hbf) {
    __shared__ float sa[8], sb[8];
    size_t row = blockIdx.x;
    int tid = threadIdx.x;
    float4 v = *(const float4*)(x2 + row * Cd + tid * 4);
    float s = v.x + v.y + v.z + v.w;
    float q = v.x * v.x + v.y * v.y + v.z * v.z + v.w * v.w;
    block_reduce2(s, q, sa, sb);
    float mu = s * (1.f / Cd);
    float rs = rsqrtf(q * (1.f / Cd) - mu * mu + 1e-5f);
    float4 gv = *(const float4*)(g2 + tid * 4);
    float4 bv = *(const float4*)(b2 + tid * 4);
    float4 t;
    t.x = (v.x - mu) * rs * gv.x + bv.x;
    t.y = (v.y - mu) * rs * gv.y + bv.y;
    t.z = (v.z - mu) * rs * gv.z + bv.z;
    t.w = (v.w - mu) * rs * gv.w + bv.w;
    *(float4*)(x3 + row * Cd + tid * 4) = t;
    float s2 = t.x + t.y + t.z + t.w;
    float q2 = t.x * t.x + t.y * t.y + t.z * t.z + t.w * t.w;
    block_reduce2(s2, q2, sa, sb);
    float mu2 = s2 * (1.f / Cd);
    float rs2 = rsqrtf(q2 * (1.f / Cd) - mu2 * mu2 + 1e-5f);
    float4 gv2 = *(const float4*)(gf + tid * 4);
    float4 bv2 = *(const float4*)(bfp + tid * 4);
    float4 o;
    o.x = (t.x - mu2) * rs2 * gv2.x + bv2.x;
    o.y = (t.y - mu2) * rs2 * gv2.y + bv2.y;
    o.z = (t.z - mu2) * rs2 * gv2.z + bv2.z;
    o.w = (t.w - mu2) * rs2 * gv2.w + bv2.w;
    *(float4*)(hf32 + row * Cd + tid * 4) = o;
    v4bf ob;
    ob[0] = (__bf16)o.x; ob[1] = (__bf16)o.y; ob[2] = (__bf16)o.z; ob[3] = (__bf16)o.w;
    *(v4bf*)(hbf + row * Cd + tid * 4) = ob;
}

// ---------------- QKV projections (fused, one A load feeds 3 WMMAs) ----------

__global__ __launch_bounds__(128) void qkv_kernel(
        const __bf16* __restrict__ hb,                        // [B*T, C]
        const __bf16* __restrict__ wq, const __bf16* __restrict__ wk,
        const __bf16* __restrict__ wv,                        // [H, C, HD]
        const float* __restrict__ bq, const float* __restrict__ bk,
        const float* __restrict__ bv,                         // [H, HD]
        __bf16* __restrict__ qo,                              // [B,H,T,HD]
        __bf16* __restrict__ kTo,                             // [B,H,HD,T]
        __bf16* __restrict__ vo) {                            // [B,H,T,HD]
    int lane = threadIdx.x & 31;
    int wvid = threadIdx.x >> 5;
    int t0 = blockIdx.x * 16;
    int hh = blockIdx.y;
    int b  = blockIdx.z;
    int n0 = wvid * 16;

    const __bf16* A  = hb + (size_t)(b * Td + t0) * Cd;
    const __bf16* Bq = wq + (size_t)hh * Cd * HDd;
    const __bf16* Bk = wk + (size_t)hh * Cd * HDd;
    const __bf16* Bv = wv + (size_t)hh * Cd * HDd;

    v8f aq = {}, ak = {}, av = {};
    for (int k0 = 0; k0 < Cd; k0 += 32) {
        v16bf a  = load_a_tile(A, Cd, 0, k0, lane);
        v16bf b0 = load_b_tile(Bq, HDd, k0, n0, lane);
        v16bf b1 = load_b_tile(Bk, HDd, k0, n0, lane);
        v16bf b2 = load_b_tile(Bv, HDd, k0, n0, lane);
        aq = wmma_bf16(a, b0, aq);
        ak = wmma_bf16(a, b1, ak);
        av = wmma_bf16(a, b2, av);
    }
    int col = n0 + (lane & 15);
    int rb  = (lane >> 4) * 8;
    float biasq = bq[hh * HDd + col];
    float biask = bk[hh * HDd + col];
    float biasv = bv[hh * HDd + col];
    size_t bh = (size_t)(b * Hh + hh);
    size_t qbase = (bh * Td + t0 + rb) * HDd + col;
    size_t kbase = (bh * HDd + col) * Td + (t0 + rb);
#pragma unroll
    for (int r = 0; r < 8; r++) {
        qo[qbase + (size_t)r * HDd] = (__bf16)(aq[r] + biasq);
        vo[qbase + (size_t)r * HDd] = (__bf16)(av[r] + biasv);
        kTo[kbase + r]              = (__bf16)(ak[r] + biask);
    }
}

// ---------------- Flash attention (causal) + residual x2 = h + o -------------

#define ATT_WAVES 4

__global__ __launch_bounds__(128) void attn_kernel(
        const __bf16* __restrict__ q,   // [B,H,T,HD]
        const __bf16* __restrict__ kT,  // [B,H,HD,T]
        const __bf16* __restrict__ v,   // [B,H,T,HD]
        const float* __restrict__ h,    // [B,T,C]
        float* __restrict__ x2) {       // [B,T,C]
    __shared__ __bf16 pl[ATT_WAVES][16 * 32];
    int lane = threadIdx.x & 31;
    int wvid = threadIdx.x >> 5;
    int tile = blockIdx.x * ATT_WAVES + wvid;
    int hh = blockIdx.y, b = blockIdx.z;
    int t0 = tile * 16;
    size_t bh = (size_t)(b * Hh + hh);
    const __bf16* qp = q  + bh * Td * HDd;
    const __bf16* kp = kT + bh * HDd * Td;
    const __bf16* vp = v  + bh * Td * HDd;

    v16bf qa0 = load_a_tile(qp, HDd, t0, 0, lane);
    v16bf qa1 = load_a_tile(qp, HDd, t0, 32, lane);

    v8f o0 = {}, o1 = {}, o2 = {}, o3 = {};
    float m[8], l[8];
#pragma unroll
    for (int r = 0; r < 8; r++) { m[r] = -__builtin_inff(); l[r] = 0.f; }

    const float scale = 0.03125f;            // 1/sqrt(C) = 1/32
    int rowoff = (lane >> 4) * 8;
    int lcol = lane & 15;
    __bf16* lp = pl[wvid];

    for (int s0 = 0; s0 <= t0 + 15; s0 += 32) {
        v8f sA = {}, sB = {};
        sA = wmma_bf16(qa0, load_b_tile(kp, Td, 0,  s0,      lane), sA);
        sA = wmma_bf16(qa1, load_b_tile(kp, Td, 32, s0,      lane), sA);
        sB = wmma_bf16(qa0, load_b_tile(kp, Td, 0,  s0 + 16, lane), sB);
        sB = wmma_bf16(qa1, load_b_tile(kp, Td, 32, s0 + 16, lane), sB);

        float p0[8], p1[8];
#pragma unroll
        for (int r = 0; r < 8; r++) {
            int trow = t0 + rowoff + r;
            float a = sA[r] * scale;
            float c = sB[r] * scale;
            if (s0 + lcol > trow)      a = -__builtin_inff();
            if (s0 + 16 + lcol > trow) c = -__builtin_inff();
            float mx = fmaxf(a, c);
#pragma unroll
            for (int off = 1; off < 16; off <<= 1) mx = fmaxf(mx, __shfl_xor(mx, off));
            float mn = fmaxf(m[r], mx);
            float corr = __expf(m[r] - mn);
            float ea = __expf(a - mn);
            float ec = __expf(c - mn);
            float rs = ea + ec;
#pragma unroll
            for (int off = 1; off < 16; off <<= 1) rs += __shfl_xor(rs, off);
            l[r] = l[r] * corr + rs;
            m[r] = mn;
            o0[r] *= corr; o1[r] *= corr; o2[r] *= corr; o3[r] *= corr;
            p0[r] = ea; p1[r] = ec;
        }

        // transpose P from C/D layout to A layout through LDS
#pragma unroll
        for (int r = 0; r < 8; r++) {
            int rl = rowoff + r;
            lp[rl * 32 + lcol]      = (__bf16)p0[r];
            lp[rl * 32 + 16 + lcol] = (__bf16)p1[r];
        }
        asm volatile("s_wait_dscnt 0" ::: "memory");
        v16bf pa;
        {
            int row = lane & 15;
            int koff = (lane >> 4) << 3;
            const __bf16* pp = lp + row * 32 + koff;
            union { v16bf vv; v8bf hh2[2]; } u;
            u.hh2[0] = *(const v8bf*)(pp);
            u.hh2[1] = *(const v8bf*)(pp + 16);
            pa = u.vv;
        }

        o0 = wmma_bf16(pa, load_b_tile(vp, HDd, s0, 0,  lane), o0);
        o1 = wmma_bf16(pa, load_b_tile(vp, HDd, s0, 16, lane), o1);
        o2 = wmma_bf16(pa, load_b_tile(vp, HDd, s0, 32, lane), o2);
        o3 = wmma_bf16(pa, load_b_tile(vp, HDd, s0, 48, lane), o3);
    }

    size_t rowbase = (size_t)b * Td + t0 + rowoff;
    int cbase = hh * HDd;
#pragma unroll
    for (int r = 0; r < 8; r++) {
        float inv = 1.f / l[r];
        size_t i0 = (rowbase + r) * Cd + cbase;
        x2[i0 + lcol]      = h[i0 + lcol]      + o0[r] * inv;
        x2[i0 + 16 + lcol] = h[i0 + 16 + lcol] + o1[r] * inv;
        x2[i0 + 32 + lcol] = h[i0 + 32 + lcol] + o2[r] * inv;
        x2[i0 + 48 + lcol] = h[i0 + 48 + lcol] + o3[r] * inv;
    }
}

// ---------------- FFN GEMMs: TDM-staged, double-buffered LDS ----------------
// Workgroup (8 waves) computes a 64x128 macro-tile of C = A[M,K] @ B[K,NLD].
// Wave 0 issues tensor_load_to_lds DMAs for the next K-chunk (64 deep) while
// all 8 waves run WMMAs from the current LDS buffers.
// EPI 0: Y = relu(acc + bias) -> bf16.   EPI 1: Y = acc + bias + add0 + add1.
template <int KTOT, int NLD, int EPI>
__global__ __launch_bounds__(256) void gemm_tdm_kernel(
        const __bf16* __restrict__ A,     // [M, KTOT]
        const __bf16* __restrict__ Bm,    // [KTOT, NLD]
        const float* __restrict__ bias,   // [NLD]
        const float* __restrict__ add0,   // EPI1 only
        const float* __restrict__ add1,   // EPI1 only
        __bf16* __restrict__ Ybf,         // EPI0 output
        float* __restrict__ Yf) {         // EPI1 output
    __shared__ __bf16 Ab[2][64 * 64];     // 64 rows x 64 k      (8 KB each)
    __shared__ __bf16 Bb[2][64 * 128];    // 64 k    x 128 cols  (16 KB each)

    int lane = threadIdx.x & 31;
    int w = threadIdx.x >> 5;             // 8 waves: 4 M-tiles x 2 N-halves
    int mw = (w >> 1) * 16;
    int nw = (w & 1) * 64;
    int Mwg = blockIdx.x * 64;
    int Nwg = blockIdx.y * 128;

    constexpr int NCH = KTOT / 64;
    v8f acc0 = {}, acc1 = {}, acc2 = {}, acc3 = {};

    auto stage = [&](int buf, int k0) {
        tdm_load_2d((unsigned)(size_t)&Ab[buf][0],
                    A + (size_t)Mwg * KTOT + k0, 64, 64, KTOT);
        tdm_load_2d((unsigned)(size_t)&Bb[buf][0],
                    Bm + (size_t)k0 * NLD + Nwg, 128, 64, NLD);
    };

    if (w == 0) {
        stage(0, 0);
        __builtin_amdgcn_s_wait_tensorcnt(0);
    }
    __syncthreads();

    for (int i = 0; i < NCH; ++i) {
        int cur = i & 1;
        if (w == 0 && i + 1 < NCH) stage(cur ^ 1, (i + 1) * 64);

        const __bf16* ap = &Ab[cur][0];
        const __bf16* bp = &Bb[cur][0];
#pragma unroll
        for (int ks = 0; ks < 64; ks += 32) {
            v16bf a = load_a_tile(ap, 64, mw, ks, lane);
            acc0 = wmma_bf16(a, load_b_tile(bp, 128, ks, nw +  0, lane), acc0);
            acc1 = wmma_bf16(a, load_b_tile(bp, 128, ks, nw + 16, lane), acc1);
            acc2 = wmma_bf16(a, load_b_tile(bp, 128, ks, nw + 32, lane), acc2);
            acc3 = wmma_bf16(a, load_b_tile(bp, 128, ks, nw + 48, lane), acc3);
        }
        asm volatile("s_wait_dscnt 0" ::: "memory");
        if (w == 0) __builtin_amdgcn_s_wait_tensorcnt(0);
        __syncthreads();
    }

    int colb = Nwg + nw + (lane & 15);
    int rb = Mwg + mw + ((lane >> 4) << 3);
    v8f accs[4] = {acc0, acc1, acc2, acc3};
#pragma unroll
    for (int j = 0; j < 4; j++) {
        int col = colb + j * 16;
        float bs = bias[col];
#pragma unroll
        for (int r = 0; r < 8; r++) {
            float vv = accs[j][r] + bs;
            size_t idx = (size_t)(rb + r) * NLD + col;
            if constexpr (EPI == 0) {
                Ybf[idx] = (__bf16)fmaxf(vv, 0.f);
            } else {
                Yf[idx] = vv + add0[idx] + add1[idx];
            }
        }
    }
}

// ---------------- host orchestration ----------------

extern "C" void kernel_launch(void* const* d_in, const int* in_sizes, int n_in,
                              void* d_out, int out_size, void* d_ws, size_t ws_size,
                              hipStream_t stream) {
    (void)in_sizes; (void)n_in; (void)out_size; (void)ws_size;
    const float* x   = (const float*)d_in[0];
    const float* Wq  = (const float*)d_in[1];
    const float* bq  = (const float*)d_in[2];
    const float* Wk  = (const float*)d_in[3];
    const float* bk  = (const float*)d_in[4];
    const float* Wv  = (const float*)d_in[5];
    const float* bv  = (const float*)d_in[6];
    const float* g1  = (const float*)d_in[7];
    const float* b1  = (const float*)d_in[8];
    const float* g2  = (const float*)d_in[9];
    const float* b2  = (const float*)d_in[10];
    const float* gf  = (const float*)d_in[11];
    const float* bf_ = (const float*)d_in[12];
    const float* W1  = (const float*)d_in[13];
    const float* b1f = (const float*)d_in[14];
    const float* W2  = (const float*)d_in[15];
    const float* b2f = (const float*)d_in[16];
    float* out = (float*)d_out;

    char* w = (char*)d_ws;
    auto take = [&](size_t bytes) {
        char* p = w;
        w += (bytes + 255) & ~(size_t)255;
        return p;
    };
    float*  h_f32   = (float*) take((size_t)Rows * Cd * 4);
    float*  x2_f32  = (float*) take((size_t)Rows * Cd * 4);
    float*  x3_f32  = (float*) take((size_t)Rows * Cd * 4);
    float*  hf_f32  = (float*) take((size_t)Rows * Cd * 4);
    __bf16* h_bf    = (__bf16*)take((size_t)Rows * Cd * 2);
    __bf16* hf_bf   = (__bf16*)take((size_t)Rows * Cd * 2);
    __bf16* q_bf    = (__bf16*)take((size_t)Bd * Hh * Td * HDd * 2);
    __bf16* kT_bf   = (__bf16*)take((size_t)Bd * Hh * Td * HDd * 2);
    __bf16* v_bf    = (__bf16*)take((size_t)Bd * Hh * Td * HDd * 2);
    __bf16* Wq_bf   = (__bf16*)take((size_t)Hh * Cd * HDd * 2);
    __bf16* Wk_bf   = (__bf16*)take((size_t)Hh * Cd * HDd * 2);
    __bf16* Wv_bf   = (__bf16*)take((size_t)Hh * Cd * HDd * 2);
    __bf16* W1_bf   = (__bf16*)take((size_t)Cd * Fd * 2);
    __bf16* W2_bf   = (__bf16*)take((size_t)Fd * Cd * 2);
    __bf16* y1_bf   = (__bf16*)take((size_t)Rows * Fd * 2);

    cvt_bf16_kernel<<<Hh * Cd * HDd / 1024, 256, 0, stream>>>(Wq, Wq_bf);
    cvt_bf16_kernel<<<Hh * Cd * HDd / 1024, 256, 0, stream>>>(Wk, Wk_bf);
    cvt_bf16_kernel<<<Hh * Cd * HDd / 1024, 256, 0, stream>>>(Wv, Wv_bf);
    cvt_bf16_kernel<<<Cd * Fd / 1024, 256, 0, stream>>>(W1, W1_bf);
    cvt_bf16_kernel<<<Fd * Cd / 1024, 256, 0, stream>>>(W2, W2_bf);

    ln_kernel<<<Rows, 256, 0, stream>>>(x, g1, b1, h_f32, h_bf);

    qkv_kernel<<<dim3(Td / 16, Hh, Bd), 128, 0, stream>>>(
        h_bf, Wq_bf, Wk_bf, Wv_bf, bq, bk, bv, q_bf, kT_bf, v_bf);

    attn_kernel<<<dim3(Td / 16 / ATT_WAVES, Hh, Bd), 32 * ATT_WAVES, 0, stream>>>(
        q_bf, kT_bf, v_bf, h_f32, x2_f32);

    ln2f_kernel<<<Rows, 256, 0, stream>>>(x2_f32, g2, b2, gf, bf_, x3_f32, hf_f32, hf_bf);

    // FFN1: y1 = relu(hf @ W1 + b1f)        [4096,1024] x [1024,4096]
    gemm_tdm_kernel<Cd, Fd, 0><<<dim3(Rows / 64, Fd / 128), 256, 0, stream>>>(
        hf_bf, W1_bf, b1f, nullptr, nullptr, y1_bf, nullptr);
    // FFN2: out = x3 + hf + (y1 @ W2 + b2f) [4096,4096] x [4096,1024]
    gemm_tdm_kernel<Fd, Cd, 1><<<dim3(Rows / 64, Cd / 128), 256, 0, stream>>>(
        y1_bf, W2_bf, b2f, x3_f32, hf_f32, nullptr, out);
}
